// SSM_spa_1305670058664
// MI455X (gfx1250) — matrix operand order repaired and checked by reading, not compiled, not run on hardware
//
#include <hip/hip_runtime.h>

// ---------------------------------------------------------------------------
// Mamba SSM block for MI455X (gfx1250, wave32, WMMA).
// GEMMs run as v_wmma_f32_16x16x32_bf16 with f32 accumulation.
// ---------------------------------------------------------------------------

#define B_SZ     16
#define SEQ_L    1024
#define D_MODEL  1024
#define D_INNER  1024
#define TWO_DI   2048
#define DT_RANK  64
#define NROWS    (B_SZ * SEQ_L)        // 16384 GEMM rows
#define XP_NPAD  128                   // x_proj N (68) padded to 128

typedef __attribute__((ext_vector_type(16))) __bf16 v16bf;
typedef __attribute__((ext_vector_type(8)))  float  v8f;

union Frag { uint4 q[2]; v16bf v; };   // 32 bytes = 16 bf16

__device__ __forceinline__ unsigned short f2bf(float f) {
  unsigned u = __float_as_uint(f);
  u += 0x7FFFu + ((u >> 16) & 1u);     // round-to-nearest-even
  return (unsigned short)(u >> 16);
}
__device__ __forceinline__ float bf2f(unsigned short h) {
  return __uint_as_float(((unsigned)h) << 16);
}
__device__ __forceinline__ float sigmoidf_(float x) {
  return 1.0f / (1.0f + __expf(-x));
}

// ---------------------------------------------------------------------------
// Weight transpose + f32->bf16: W (K x N, row major) -> Wt (Npad x K, row
// major, zero padded). Makes B-fragment loads contiguous b128 per lane.
// ---------------------------------------------------------------------------
__global__ void transpose_w_bf16(const float* __restrict__ W,
                                 unsigned short* __restrict__ Wt,
                                 int K, int N, int Npad) {
  int tid = blockIdx.x * blockDim.x + threadIdx.x;
  if (tid >= Npad * K) return;
  int n = tid / K;
  int k = tid - n * K;
  float v = (n < N) ? W[(size_t)k * N + n] : 0.0f;
  Wt[(size_t)n * K + k] = f2bf(v);
}

__global__ void f32_to_bf16(const float* __restrict__ in,
                            unsigned short* __restrict__ out, int n) {
  int tid = blockIdx.x * blockDim.x + threadIdx.x;
  if (tid < n) out[tid] = f2bf(in[tid]);
}

// ---------------------------------------------------------------------------
// WMMA GEMM: C[M x N] = A[M x K] * Bt[N x K]^T  (A,Bt bf16; C f32 or bf16)
// Block = 128 threads = 4 waves; wave tile 64x64 (4x4 WMMA 16x16x32 tiles).
// grid = (N/128, M/128). All dims multiples of the tile sizes (padded).
// ---------------------------------------------------------------------------
template <bool OUT_BF16>
__global__ __launch_bounds__(128)
void gemm_wmma_bf16(const unsigned short* __restrict__ A, int lda,
                    const unsigned short* __restrict__ Bt, int ldb,
                    void* __restrict__ Cv, int ldc, int K) {
  const int lane  = threadIdx.x & 31;
  const int wave  = threadIdx.x >> 5;
  const int lrow  = lane & 15;     // row-within-tile for A, col for B/C
  const int lhalf = lane >> 4;     // half-wave select
  const int bm = blockIdx.y * 128 + (wave >> 1) * 64;
  const int bn = blockIdx.x * 128 + (wave & 1) * 64;

  v8f acc[4][4];
#pragma unroll
  for (int i = 0; i < 4; ++i)
#pragma unroll
    for (int j = 0; j < 4; ++j)
#pragma unroll
      for (int r = 0; r < 8; ++r) acc[i][j][r] = 0.0f;

  // A fragment base: lane holds row (tile_row + lrow); half-waves split K
  // 0-15 / 16-31, each lane's 16 K-values contiguous.
  const unsigned short* arow[4];
  const unsigned short* brow[4];
#pragma unroll
  for (int i = 0; i < 4; ++i)
    arow[i] = A + (size_t)(bm + i * 16 + lrow) * lda + lhalf * 16;
  // B fragment base: lane holds column (tile_col + lrow); half-waves split
  // K 0-15 / 16-31, each lane's 16 K-values contiguous in transposed weight.
#pragma unroll
  for (int j = 0; j < 4; ++j)
    brow[j] = Bt + (size_t)(bn + j * 16 + lrow) * ldb + lhalf * 16;

  for (int k0 = 0; k0 < K; k0 += 32) {
    Frag a[4], b[4];
#pragma unroll
    for (int i = 0; i < 4; ++i) {
      a[i].q[0] = *(const uint4*)(arow[i] + k0);
      a[i].q[1] = *(const uint4*)(arow[i] + k0 + 8);
    }
#pragma unroll
    for (int j = 0; j < 4; ++j) {
      b[j].q[0] = *(const uint4*)(brow[j] + k0);
      b[j].q[1] = *(const uint4*)(brow[j] + k0 + 8);
    }
#pragma unroll
    for (int i = 0; i < 4; ++i)
#pragma unroll
      for (int j = 0; j < 4; ++j)
        acc[i][j] = __builtin_amdgcn_wmma_f32_16x16x32_bf16(
            false, a[i].v, false, b[j].v, (short)0, acc[i][j], false, false);
  }

  // Epilogue: C/D layout — VGPR r: lanes0-15 -> M=r, lanes16-31 -> M=8+r.
#pragma unroll
  for (int i = 0; i < 4; ++i) {
    const int r0 = bm + i * 16 + lhalf * 8;
#pragma unroll
    for (int j = 0; j < 4; ++j) {
      const int col = bn + j * 16 + lrow;
#pragma unroll
      for (int r = 0; r < 8; ++r) {
        if constexpr (OUT_BF16)
          ((unsigned short*)Cv)[(size_t)(r0 + r) * ldc + col] = f2bf(acc[i][j][r]);
        else
          ((float*)Cv)[(size_t)(r0 + r) * ldc + col] = acc[i][j][r];
      }
    }
  }
}

// ---------------------------------------------------------------------------
// Causal depthwise conv (k=4) along L + SiLU; reads xi half of xz (bf16),
// writes bf16 activations for the x_proj GEMM / scan.
// ---------------------------------------------------------------------------
__global__ void conv_silu(const unsigned short* __restrict__ xz,
                          const float* __restrict__ cw,
                          const float* __restrict__ cb,
                          unsigned short* __restrict__ xc) {
  int tid = blockIdx.x * blockDim.x + threadIdx.x;  // < B*L*D
  int d = tid & (D_INNER - 1);
  int t = (tid >> 10) & (SEQ_L - 1);
  int b = tid >> 20;
  const unsigned short* base = xz + ((size_t)b << 10) * TWO_DI + d;
  float acc = cb[d];
#pragma unroll
  for (int k = 0; k < 4; ++k) {
    int tt = t - 3 + k;
    if (tt >= 0) acc += bf2f(base[(size_t)tt * TWO_DI]) * cw[d * 4 + k];
  }
  acc = acc * sigmoidf_(acc);
  xc[tid] = f2bf(acc);
}

// Extract dt columns (0..63) from dbl (ld=128) to packed bf16 (ld=64)
__global__ void extract_dt(const float* __restrict__ dbl,
                           unsigned short* __restrict__ dt) {
  int tid = blockIdx.x * blockDim.x + threadIdx.x;  // < NROWS*64
  int r = tid >> 6;
  int c = tid & 63;
  dt[tid] = f2bf(dbl[(size_t)r * XP_NPAD + c]);
}

// delta = softplus(delta_raw + dt_proj_b)
__global__ void bias_softplus(float* __restrict__ delta,
                              const float* __restrict__ bias) {
  int tid = blockIdx.x * blockDim.x + threadIdx.x;  // < NROWS*D_INNER
  int d = tid & (D_INNER - 1);
  float x = delta[tid] + bias[d];
  delta[tid] = (x > 20.0f) ? x : log1pf(__expf(x));
}

// ---------------------------------------------------------------------------
// Sequential selective scan over L (d_state = 2), one thread per (b, d)
// channel; d contiguous within a block for coalesced access. Fuses the
// D-skip and SiLU(res) gate, emits bf16 rows for the out_proj GEMM.
// ---------------------------------------------------------------------------
__global__ void ssm_scan(const float* __restrict__ delta,
                         const float* __restrict__ dbl,
                         const unsigned short* __restrict__ xc,
                         const unsigned short* __restrict__ xz,
                         const float* __restrict__ A_log,
                         const float* __restrict__ Dsk,
                         unsigned short* __restrict__ y) {
  int tid = blockIdx.x * blockDim.x + threadIdx.x;  // < B*D
  int d = tid & (D_INNER - 1);
  int b = tid >> 10;
  float A0 = -__expf(A_log[d * 2 + 0]);
  float A1 = -__expf(A_log[d * 2 + 1]);
  float Dv = Dsk[d];
  float h0 = 0.0f, h1 = 0.0f;
  size_t rowbase = (size_t)b << 10;
  for (int t = 0; t < SEQ_L; ++t) {
    size_t row = rowbase + t;
    float dl = delta[row * D_INNER + d];
    const float* bc = dbl + row * XP_NPAD + DT_RANK;  // B0 B1 C0 C1
    float B0 = bc[0], B1 = bc[1], C0 = bc[2], C1 = bc[3];
    float xi = bf2f(xc[row * D_INNER + d]);
    float dbu = dl * xi;
    h0 = __expf(dl * A0) * h0 + dbu * B0;
    h1 = __expf(dl * A1) * h1 + dbu * B1;
    float yv = h0 * C0 + h1 * C1 + xi * Dv;
    float rs = bf2f(xz[row * TWO_DI + D_INNER + d]);
    yv *= rs * sigmoidf_(rs);
    y[row * D_INNER + d] = f2bf(yv);
  }
}

// ---------------------------------------------------------------------------
// LayerNorm over out_c = 1024, one block per row, in place on d_out.
// ---------------------------------------------------------------------------
__global__ __launch_bounds__(256)
void layernorm_rows(float* __restrict__ out,
                    const float* __restrict__ w,
                    const float* __restrict__ b) {
  __shared__ float ss[256], sq[256];
  float* rp = out + (size_t)blockIdx.x * 1024;
  float v[4], s = 0.0f, q = 0.0f;
#pragma unroll
  for (int k = 0; k < 4; ++k) {
    v[k] = rp[threadIdx.x + k * 256];
    s += v[k];
    q += v[k] * v[k];
  }
  ss[threadIdx.x] = s;
  sq[threadIdx.x] = q;
  __syncthreads();
  for (int off = 128; off > 0; off >>= 1) {
    if ((int)threadIdx.x < off) {
      ss[threadIdx.x] += ss[threadIdx.x + off];
      sq[threadIdx.x] += sq[threadIdx.x + off];
    }
    __syncthreads();
  }
  float mu = ss[0] * (1.0f / 1024.0f);
  float var = sq[0] * (1.0f / 1024.0f) - mu * mu;
  float rstd = rsqrtf(var + 1e-5f);
#pragma unroll
  for (int k = 0; k < 4; ++k) {
    int c = threadIdx.x + k * 256;
    rp[c] = (v[k] - mu) * rstd * w[c] + b[c];
  }
}

// ---------------------------------------------------------------------------
// Launcher
// ---------------------------------------------------------------------------
extern "C" void kernel_launch(void* const* d_in, const int* in_sizes, int n_in,
                              void* d_out, int out_size, void* d_ws,
                              size_t ws_size, hipStream_t stream) {
  const float* x        = (const float*)d_in[0];
  const float* in_w     = (const float*)d_in[1];
  const float* conv_w   = (const float*)d_in[2];
  const float* conv_b   = (const float*)d_in[3];
  const float* xproj_w  = (const float*)d_in[4];
  const float* dtproj_w = (const float*)d_in[5];
  const float* dtproj_b = (const float*)d_in[6];
  const float* A_log    = (const float*)d_in[7];
  const float* D_skip   = (const float*)d_in[8];
  const float* out_w    = (const float*)d_in[9];
  const float* ln_w     = (const float*)d_in[10];
  const float* ln_b     = (const float*)d_in[11];

  char* ws = (char*)d_ws;
  size_t off = 0;
  auto take = [&](size_t bytes) {
    void* p = ws + off;
    off += (bytes + 255) & ~(size_t)255;
    return p;
  };
  unsigned short* U    = (unsigned short*)take((size_t)NROWS * D_MODEL * 2);   // 32 MB
  unsigned short* WtI  = (unsigned short*)take((size_t)TWO_DI * D_MODEL * 2);  //  4 MB
  unsigned short* XZ   = (unsigned short*)take((size_t)NROWS * TWO_DI * 2);    // 64 MB
  unsigned short* XC   = (unsigned short*)take((size_t)NROWS * D_INNER * 2);   // 32 MB
  unsigned short* WtX  = (unsigned short*)take((size_t)XP_NPAD * D_INNER * 2); // 256 KB
  float*          DBL  = (float*)take((size_t)NROWS * XP_NPAD * 4);            //  8 MB
  unsigned short* DT   = (unsigned short*)take((size_t)NROWS * DT_RANK * 2);   //  2 MB
  unsigned short* WtD  = (unsigned short*)take((size_t)D_INNER * DT_RANK * 2); // 128 KB
  float*          DELTA= (float*)take((size_t)NROWS * D_INNER * 4);            // 64 MB
  unsigned short* Y    = (unsigned short*)take((size_t)NROWS * D_INNER * 2);   // 32 MB
  unsigned short* WtO  = (unsigned short*)take((size_t)D_MODEL * D_MODEL * 2); //  2 MB

  const int T = 256;
  // --- stage weights (transpose + bf16) ---
  transpose_w_bf16<<<(TWO_DI * D_MODEL + T - 1) / T, T, 0, stream>>>(
      in_w, WtI, D_MODEL, TWO_DI, TWO_DI);
  transpose_w_bf16<<<(XP_NPAD * D_INNER + T - 1) / T, T, 0, stream>>>(
      xproj_w, WtX, D_INNER, DT_RANK + 4, XP_NPAD);
  transpose_w_bf16<<<(D_INNER * DT_RANK + T - 1) / T, T, 0, stream>>>(
      dtproj_w, WtD, DT_RANK, D_INNER, D_INNER);
  transpose_w_bf16<<<(D_MODEL * D_MODEL + T - 1) / T, T, 0, stream>>>(
      out_w, WtO, D_INNER, D_MODEL, D_MODEL);

  // --- stage activations ---
  f32_to_bf16<<<(NROWS * D_MODEL + T - 1) / T, T, 0, stream>>>(
      x, U, NROWS * D_MODEL);

  // --- in_proj GEMM: (16384x1024)x(1024x2048) -> bf16 xz ---
  gemm_wmma_bf16<true><<<dim3(TWO_DI / 128, NROWS / 128), 128, 0, stream>>>(
      U, D_MODEL, WtI, D_MODEL, XZ, TWO_DI, D_MODEL);

  // --- causal conv + SiLU ---
  conv_silu<<<(NROWS * D_INNER + T - 1) / T, T, 0, stream>>>(
      XZ, conv_w, conv_b, XC);

  // --- x_proj GEMM: (16384x1024)x(1024x128pad) -> f32 dbl ---
  gemm_wmma_bf16<false><<<dim3(XP_NPAD / 128, NROWS / 128), 128, 0, stream>>>(
      XC, D_INNER, WtX, D_INNER, DBL, XP_NPAD, D_INNER);

  // --- dt extraction + dt_proj GEMM: (16384x64)x(64x1024) -> f32 delta ---
  extract_dt<<<(NROWS * DT_RANK + T - 1) / T, T, 0, stream>>>(DBL, DT);
  gemm_wmma_bf16<false><<<dim3(D_INNER / 128, NROWS / 128), 128, 0, stream>>>(
      DT, DT_RANK, WtD, DT_RANK, DELTA, D_INNER, DT_RANK);
  bias_softplus<<<(NROWS * D_INNER + T - 1) / T, T, 0, stream>>>(DELTA, dtproj_b);

  // --- selective scan (sequential in L) + skip + gate -> bf16 y ---
  ssm_scan<<<(B_SZ * D_INNER + T - 1) / T, T, 0, stream>>>(
      DELTA, DBL, XC, XZ, A_log, D_skip, Y);

  // --- out_proj GEMM: (16384x1024)x(1024x1024) -> f32 d_out ---
  gemm_wmma_bf16<false><<<dim3(D_MODEL / 128, NROWS / 128), 128, 0, stream>>>(
      Y, D_INNER, WtO, D_INNER, d_out, D_MODEL, D_INNER);

  // --- LayerNorm in place ---
  layernorm_rows<<<NROWS, 256, 0, stream>>>((float*)d_out, ln_w, ln_b);
}
